// PerformerDecoder_38998303048009
// MI455X (gfx1250) — matrix operand order repaired and checked by reading, not compiled, not run on hardware
//
#include <hip/hip_runtime.h>
#include <hip/hip_bf16.h>

// ---------------- types ----------------
typedef __attribute__((ext_vector_type(16))) __bf16 v16bf;
typedef __attribute__((ext_vector_type(8)))  float  v8f;

// ---------------- model constants ----------------
#define BATCH  8
#define TLEN   4096
#define NROWS  (BATCH * TLEN)      // 32768
#define HID    256
#define CDIM   128
#define PHID   64
#define HEADS  2
#define DH     32
#define BHN    (BATCH * HEADS)     // 16
#define NBV    110
#define NBP    128                 // padded feature count (mult of 32)
#define NQROWS (BHN * TLEN)        // 65536

#define GF_BIAS 1
#define GF_ACC  2
#define GF_RELU 4
#define GF_GELU 8
#define GF_TRA  16
#define GF_TRB  32

// =======================================================================
// Generic bf16 WMMA GEMM:  C[M,N] (+)= op(A) * op(B) (+ bias) (epilogue)
//   !TRA: A is [M,K] row-major (lda = row stride)   TRA: A is [K,M]
//   !TRB: B is [N,K] row-major (weights)            TRB: B is [K,N]
// block = 128 threads (4 waves); block tile 64(M) x BN(N); K-step 32.
// wave wv owns rows [wv*16, wv*16+16) x all BN cols -> BN/16 wmma / K-step,
// all sharing one A fragment (A-operand reuse).
// M mult of 64, N mult of BN, K mult of 32 (guaranteed by caller).
// All operand base pointers / strides are 16B aligned (guaranteed).
// =======================================================================
template<bool TRA, bool TRB, int BN>
__global__ __launch_bounds__(128)
void k_gemm(const __bf16* __restrict__ A, const __bf16* __restrict__ B,
            const float* __restrict__ bias, float* __restrict__ C,
            __bf16* __restrict__ Cbf,
            int K, int Nn, int lda, int ldb,
            long sA, long sB, long sC, int flags)
{
    const int NF = BN / 16;          // B fragments per wave
    __shared__ __bf16 As[64 * 32];   // [m][k]
    __shared__ __bf16 Bs[BN * 32];   // [n][k]

    const int tid  = threadIdx.x;
    const int lane = tid & 31;
    const int wv   = tid >> 5;
    const int half = lane >> 4;      // 0/1
    const int l15  = lane & 15;
    const int row0 = blockIdx.y * 64;
    const int col0 = blockIdx.x * BN;
    const long bA  = (long)blockIdx.z * sA;
    const long bB  = (long)blockIdx.z * sB;
    const long bC  = (long)blockIdx.z * sC;

    // per-lane LDS dword gather indices per ISA 7.12.2 fragment layouts
    int aidx[8], kboff[8];
#pragma unroll
    for (int v = 0; v < 8; ++v) {
        int ka   = ((v >> 2) << 4) + (half << 3) + ((v & 3) << 1); // A 16x32 bf16
        aidx[v]  = ((wv * 16 + l15) * 32 + ka) >> 1;
        kboff[v] = (((v >> 2) << 4) + (half << 3) + ((v & 3) << 1)) >> 1; // B 32x16 bf16
    }

    v8f acc[NF];
#pragma unroll
    for (int f = 0; f < NF; ++f)
        acc[f] = (v8f){0.f,0.f,0.f,0.f,0.f,0.f,0.f,0.f};

    const unsigned* Asu = (const unsigned*)As;
    const unsigned* Bsu = (const unsigned*)Bs;

    for (int k0 = 0; k0 < K; k0 += 32) {
        if (k0 + 32 < K)  // speculative prefetch of next A tile (global_prefetch_b8)
            __builtin_prefetch(A + bA + (long)(row0 + (tid >> 1)) * lda + (k0 + 32), 0, 1);

        // ---- stage A tile (64x32), 16 elems / thread, straight-line ----
        if (!TRA) {
            int m  = tid >> 1;
            int ch = (tid & 1) << 4;
            const __bf16* src = A + bA + (long)(row0 + m) * lda + k0 + ch;
            *(uint4*)(As + m * 32 + ch)     = *(const uint4*)(src);
            *(uint4*)(As + m * 32 + ch + 8) = *(const uint4*)(src + 8);
        } else {
            int kk = tid >> 2;           // 0..31
            int ch = (tid & 3) << 4;     // m offset 0,16,32,48
            const __bf16* src = A + bA + (long)(k0 + kk) * lda + row0 + ch;
            union { uint4 q; __bf16 h[8]; } u0, u1;
            u0.q = *(const uint4*)(src);
            u1.q = *(const uint4*)(src + 8);
#pragma unroll
            for (int j = 0; j < 8; ++j) {
                As[(ch + j) * 32 + kk]     = u0.h[j];
                As[(ch + 8 + j) * 32 + kk] = u1.h[j];
            }
        }
        // ---- stage B tile (BNx32), BN/4 elems / thread ----
        if (!TRB) {
            if (BN == 64) {
                int n  = tid >> 1;
                int ch = (tid & 1) << 4;
                const __bf16* src = B + bB + (long)(col0 + n) * ldb + k0 + ch;
                *(uint4*)(Bs + n * 32 + ch)     = *(const uint4*)(src);
                *(uint4*)(Bs + n * 32 + ch + 8) = *(const uint4*)(src + 8);
            } else {
                int n  = tid >> 2;
                int ch = (tid & 3) << 3;
                *(uint4*)(Bs + n * 32 + ch) =
                    *(const uint4*)(B + bB + (long)(col0 + n) * ldb + k0 + ch);
            }
        } else {
            if (BN == 64) {
                int kk = tid >> 2;
                int ch = (tid & 3) << 4;     // n offset 0,16,32,48
                const __bf16* src = B + bB + (long)(k0 + kk) * ldb + col0 + ch;
                union { uint4 q; __bf16 h[8]; } u0, u1;
                u0.q = *(const uint4*)(src);
                u1.q = *(const uint4*)(src + 8);
#pragma unroll
                for (int j = 0; j < 8; ++j) {
                    Bs[(ch + j) * 32 + kk]     = u0.h[j];
                    Bs[(ch + 8 + j) * 32 + kk] = u1.h[j];
                }
            } else {
                int kk = tid >> 2;
                int ch = (tid & 3) << 3;     // n offset 0,8,16,24
                union { uint4 q; __bf16 h[8]; } u;
                u.q = *(const uint4*)(B + bB + (long)(k0 + kk) * ldb + col0 + ch);
#pragma unroll
                for (int j = 0; j < 8; ++j) Bs[(ch + j) * 32 + kk] = u.h[j];
            }
        }
        __syncthreads();

        union { v16bf v; unsigned u[8]; } af;
#pragma unroll
        for (int v = 0; v < 8; ++v) af.u[v] = Asu[aidx[v]];
#pragma unroll
        for (int f = 0; f < NF; ++f) {
            union { v16bf v; unsigned u[8]; } bfm;
            const int nbase = (f * 16 + l15) * 16;   // dword row base
#pragma unroll
            for (int v = 0; v < 8; ++v) bfm.u[v] = Bsu[nbase + kboff[v]];
            acc[f] = __builtin_amdgcn_wmma_f32_16x16x32_bf16(
                         false, af.v, false, bfm.v, (short)0, acc[f], false, false);
        }
        __syncthreads();
    }

    // epilogue; C/D layout: lane -> col = lane&15, vgpr v -> row = v + 8*(lane>>4)
#pragma unroll
    for (int f = 0; f < NF; ++f) {
        const int col = col0 + f * 16 + l15;
#pragma unroll
        for (int v = 0; v < 8; ++v) {
            int  row = row0 + wv * 16 + (half << 3) + v;
            long idx = bC + (long)row * Nn + col;
            float r  = acc[f][v];
            if (flags & GF_BIAS) r += bias[col];
            if (flags & GF_ACC)  r += C[idx];
            if (flags & GF_RELU) r = fmaxf(r, 0.f);
            if (flags & GF_GELU) {
                float x3 = r * r * r;
                r = 0.5f * r * (1.f + tanhf(0.7978845608028654f * (r + 0.044715f * x3)));
            }
            C[idx] = r;
            if (Cbf) Cbf[idx] = (__bf16)r;
        }
    }
}

// ============================ support kernels ============================

// net[n,ch] = p[n,0..2] . W[ch,0..2] + b[ch]
__global__ void k_embed(const float* __restrict__ p, const float* __restrict__ W,
                        const float* __restrict__ b, float* __restrict__ out)
{
    int i = blockIdx.x * blockDim.x + threadIdx.x;
    if (i >= NROWS * HID) return;
    int ch = i & 255, n = i >> 8;
    const float* pp = p + (long)n * 3;
    out[i] = W[ch * 3 + 0] * pp[0] + W[ch * 3 + 1] * pp[1] + W[ch * 3 + 2] * pp[2] + b[ch];
}

// per-channel mean/var over all rows (training-mode batch stats)
__global__ void k_colstats(const float* __restrict__ x, float* mean, float* var, int rows)
{
    __shared__ float ssum[256], ssq[256];
    int ch = blockIdx.x;
    float s = 0.f, q = 0.f;
    for (int r = threadIdx.x; r < rows; r += 256) {
        float v = x[(long)r * HID + ch];
        s += v; q += v * v;
    }
    ssum[threadIdx.x] = s; ssq[threadIdx.x] = q;
    __syncthreads();
    for (int off = 128; off > 0; off >>= 1) {
        if (threadIdx.x < off) {
            ssum[threadIdx.x] += ssum[threadIdx.x + off];
            ssq[threadIdx.x]  += ssq[threadIdx.x + off];
        }
        __syncthreads();
    }
    if (threadIdx.x == 0) {
        float m = ssum[0] / rows;
        mean[ch] = m;
        var[ch]  = ssq[0] / rows - m * m;
    }
}

// conditional BN folded affine: s1[b,ch], s2[b,ch]
__global__ void k_cbn_params(const float* __restrict__ cond,
                             const float* __restrict__ gW, const float* __restrict__ gb,
                             const float* __restrict__ bW, const float* __restrict__ bb,
                             const float* __restrict__ mean, const float* __restrict__ var,
                             float* __restrict__ s1, float* __restrict__ s2)
{
    int i = blockIdx.x * blockDim.x + threadIdx.x;
    if (i >= BATCH * HID) return;
    int b = i >> 8, ch = i & 255;
    const float* cv = cond + (long)b * CDIM;
    float g = gb[ch], be = bb[ch];
    for (int j = 0; j < CDIM; ++j) {
        g  += cv[j] * gW[ch * CDIM + j];
        be += cv[j] * bW[ch * CDIM + j];
    }
    float inv = rsqrtf(var[ch] + 1e-5f);
    s1[i] = g * inv;
    s2[i] = be - mean[ch] * g * inv;
}

// bf16(relu(x*s1[b,ch] + s2[b,ch]))
__global__ void k_cbn_relu(const float* __restrict__ x, const float* __restrict__ s1,
                           const float* __restrict__ s2, __bf16* __restrict__ out)
{
    int i = blockIdx.x * blockDim.x + threadIdx.x;
    if (i >= NROWS * HID) return;
    int ch = i & 255, n = i >> 8, b = n >> 12;
    float v = x[i] * s1[b * HID + ch] + s2[b * HID + ch];
    out[i] = (__bf16)fmaxf(v, 0.f);
}

__global__ void k_cast(const float* __restrict__ in, __bf16* __restrict__ out, int n)
{
    int i = blockIdx.x * blockDim.x + threadIdx.x;
    if (i < n) out[i] = (__bf16)in[i];
}

// proj (110,32) -> bf16 padded (128,32), rows >=110 zero
__global__ void k_padproj(const float* __restrict__ proj, __bf16* __restrict__ out)
{
    int i = blockIdx.x * blockDim.x + threadIdx.x;
    if (i >= NBP * DH) return;
    int r = i >> 5;
    out[i] = (r < NBV) ? (__bf16)proj[i] : (__bf16)0.f;
}

// row-wise layernorm (+scale/shift) -> bf16
__global__ void k_ln(const float* __restrict__ y, const float* __restrict__ s,
                     const float* __restrict__ b, __bf16* __restrict__ out)
{
    int row = blockIdx.x * blockDim.x + threadIdx.x;
    if (row >= NROWS) return;
    const float* r = y + (long)row * PHID;
    float m = 0.f;
    for (int j = 0; j < PHID; ++j) m += r[j];
    m *= (1.f / PHID);
    float v = 0.f;
    for (int j = 0; j < PHID; ++j) { float d = r[j] - m; v += d * d; }
    v *= (1.f / PHID);
    float inv = rsqrtf(v + 1e-5f);
    for (int j = 0; j < PHID; ++j)
        out[(long)row * PHID + j] = (__bf16)((r[j] - m) * inv * s[j] + b[j]);
}

// (B,T,H*DH) f32 -> (bh,T,DH) bf16 * scale
__global__ void k_split(const float* __restrict__ in, __bf16* __restrict__ out, float scale)
{
    int i = blockIdx.x * blockDim.x + threadIdx.x;
    if (i >= BHN * TLEN * DH) return;
    int d = i & 31, t = (i >> 5) & 4095, bh = i >> 17, b = bh >> 1, h = bh & 1;
    float v = in[((long)b * TLEN + t) * PHID + h * DH + d] * scale;
    out[i] = (__bf16)v;
}

// diag[row] = 0.5 * sum_d xd^2
__global__ void k_diag(const __bf16* __restrict__ xd, float* __restrict__ diag)
{
    int row = blockIdx.x * blockDim.x + threadIdx.x;
    if (row >= NQROWS) return;
    float s = 0.f;
    for (int d = 0; d < DH; ++d) { float v = (float)xd[(long)row * DH + d]; s += v * v; }
    diag[row] = 0.5f * s;
}

// row max over the 110 valid feature columns
__global__ void k_rowmax(const float* __restrict__ u, float* __restrict__ rmax)
{
    int row = blockIdx.x * blockDim.x + threadIdx.x;
    if (row >= NQROWS) return;
    float m = -3.0e38f;
    const float* r = u + (long)row * NBP;
    for (int j = 0; j < NBV; ++j) m = fmaxf(m, r[j]);
    rmax[row] = m;
}

// per-bh max over T row-maxima
__global__ void k_bhmax(const float* __restrict__ rmax, float* __restrict__ bhm)
{
    __shared__ float sm[256];
    int bh = blockIdx.x;
    float m = -3.0e38f;
    for (int t = threadIdx.x; t < TLEN; t += 256) m = fmaxf(m, rmax[(long)bh * TLEN + t]);
    sm[threadIdx.x] = m;
    __syncthreads();
    for (int off = 128; off > 0; off >>= 1) {
        if (threadIdx.x < off) sm[threadIdx.x] = fmaxf(sm[threadIdx.x], sm[threadIdx.x + off]);
        __syncthreads();
    }
    if (threadIdx.x == 0) bhm[bh] = sm[0];
}

// FAVOR+ feature map: NB^-0.5 * (exp(u - diag - mx) + 1e-4); padded cols -> 0
__global__ void k_favor(const float* __restrict__ u, const float* __restrict__ diag,
                        const float* __restrict__ mx, int perRow, __bf16* __restrict__ out)
{
    int i = blockIdx.x * blockDim.x + threadIdx.x;
    if (i >= NQROWS * NBP) return;
    int m = i & 127, row = i >> 7;
    float val = 0.f;
    if (m < NBV) {
        float mm = perRow ? mx[row] : mx[row >> 12];
        val = 0.09534625892455922f * (expf(u[i] - diag[row] - mm) + 1e-4f);
    }
    out[i] = (__bf16)val;
}

// ksum[bh,m] = sum_t kp[bh,t,m]
__global__ void k_ksum(const __bf16* __restrict__ kp, float* __restrict__ ksum)
{
    int bh = blockIdx.x, m = threadIdx.x;
    float s = 0.f;
    for (int t = 0; t < TLEN; ++t) s += (float)kp[((long)bh * TLEN + t) * NBP + m];
    ksum[bh * NBP + m] = s;
}

// dinv[row] = 1 / (qp[row,:] . ksum[bh,:])
__global__ void k_dinv(const __bf16* __restrict__ qp, const float* __restrict__ ksum,
                       float* __restrict__ dinv)
{
    int row = blockIdx.x * blockDim.x + threadIdx.x;
    if (row >= NQROWS) return;
    int bh = row >> 12;
    float s = 0.f;
    for (int m = 0; m < NBP; ++m) s += (float)qp[(long)row * NBP + m] * ksum[bh * NBP + m];
    dinv[row] = 1.f / s;
}

// (bh,T,DH) * dinv -> merged (B,T,H*DH) bf16
__global__ void k_merge(const float* __restrict__ o, const float* __restrict__ dinv,
                        __bf16* __restrict__ out)
{
    int j = blockIdx.x * blockDim.x + threadIdx.x;
    if (j >= NROWS * PHID) return;
    int d = j & 31, h = (j >> 5) & 1, t = (j >> 6) & 4095, b = j >> 18;
    long r = (long)(b * HEADS + h) * TLEN + t;
    out[j] = (__bf16)(o[r * DH + d] * dinv[r]);
}

// out[n] = sum_c relu(net2[n,c]) * w[c] + b0
__global__ void k_fcout(const float* __restrict__ net2, const float* __restrict__ w,
                        const float* __restrict__ b, float* __restrict__ out)
{
    int n = blockIdx.x * blockDim.x + threadIdx.x;
    if (n >= NROWS) return;
    float s = 0.f;
    const float* r = net2 + (long)n * HID;
    for (int c = 0; c < HID; ++c) s += fmaxf(r[c], 0.f) * w[c];
    out[n] = s + b[0];
}

// =============================== host ===============================
extern "C" void kernel_launch(void* const* d_in, const int* in_sizes, int n_in,
                              void* d_out, int out_size, void* d_ws, size_t ws_size,
                              hipStream_t stream)
{
    (void)in_sizes; (void)n_in; (void)out_size; (void)ws_size;
    // ---- inputs (setup_inputs dict order) ----
    const float* p        = (const float*)d_in[0];
    const float* cond     = (const float*)d_in[2];
    const float* fc_p_W   = (const float*)d_in[3];
    const float* fc_p_b   = (const float*)d_in[4];
    const float* g0W      = (const float*)d_in[5];
    const float* g0b      = (const float*)d_in[6];
    const float* b0W      = (const float*)d_in[7];
    const float* b0b      = (const float*)d_in[8];
    const float* g1W      = (const float*)d_in[9];
    const float* g1b      = (const float*)d_in[10];
    const float* b1W      = (const float*)d_in[11];
    const float* b1b      = (const float*)d_in[12];
    const float* f0W      = (const float*)d_in[13];
    const float* f0b      = (const float*)d_in[14];
    const float* f1W      = (const float*)d_in[15];
    const float* f1b      = (const float*)d_in[16];
    const float* proj_in_W= (const float*)d_in[17];
    const float* proj_in_b= (const float*)d_in[18];
    const float* ln1_s    = (const float*)d_in[19];
    const float* ln1_b    = (const float*)d_in[20];
    const float* Wq       = (const float*)d_in[21];
    const float* Wk       = (const float*)d_in[22];
    const float* Wv       = (const float*)d_in[23];
    const float* Wo       = (const float*)d_in[24];
    const float* bo       = (const float*)d_in[25];
    const float* ln2_s    = (const float*)d_in[26];
    const float* ln2_b    = (const float*)d_in[27];
    const float* W1       = (const float*)d_in[28];
    const float* b1p      = (const float*)d_in[29];
    const float* W2       = (const float*)d_in[30];
    const float* b2p      = (const float*)d_in[31];
    const float* projW    = (const float*)d_in[32];
    const float* proj_out_W = (const float*)d_in[33];
    const float* proj_out_b = (const float*)d_in[34];
    const float* fc_out_W = (const float*)d_in[35];
    const float* fc_out_b = (const float*)d_in[36];
    float* out = (float*)d_out;

    // ---- workspace arena ----
    char* w = (char*)d_ws;
    auto alloc = [&](size_t bytes) -> void* {
        void* r = (void*)w;
        w += (bytes + 255) & ~(size_t)255;
        return r;
    };
    float*  net   = (float*)alloc((size_t)NROWS * HID * 4);      // 32MB (also net2)
    float*  hbuf  = (float*)alloc((size_t)NROWS * HID * 4);      // 32MB (also q/k/v lin)
    float*  ubuf  = (float*)alloc((size_t)NQROWS * NBP * 4);     // 32MB (also FFN h1)
    __bf16* bf1   = (__bf16*)alloc((size_t)NQROWS * NBP * 2);    // 16MB (act bf / qp / h1bf)
    __bf16* bf2   = (__bf16*)alloc((size_t)NQROWS * NBP * 2);    // 16MB (kp)
    float*  ybuf  = (float*)alloc((size_t)NROWS * PHID * 4);     // 8MB
    float*  obuf  = (float*)alloc((size_t)NQROWS * DH * 4);      // 8MB
    __bf16* xdq   = (__bf16*)alloc((size_t)NQROWS * DH * 2);     // 4MB
    __bf16* xdk   = (__bf16*)alloc((size_t)NQROWS * DH * 2);
    __bf16* vbf   = (__bf16*)alloc((size_t)NQROWS * DH * 2);
    __bf16* lnbf  = (__bf16*)alloc((size_t)NROWS * PHID * 2);
    __bf16* omg   = (__bf16*)alloc((size_t)NROWS * PHID * 2);
    float*  diagq = (float*)alloc((size_t)NQROWS * 4);
    float*  diagk = (float*)alloc((size_t)NQROWS * 4);
    float*  rmax  = (float*)alloc((size_t)NQROWS * 4);
    float*  bhm   = (float*)alloc(256);
    float*  ksum  = (float*)alloc((size_t)BHN * NBP * 4);
    float*  ctx   = (float*)alloc((size_t)BHN * NBP * DH * 4);
    __bf16* ctxbf = (__bf16*)alloc((size_t)BHN * NBP * DH * 2);
    float*  dinv  = (float*)alloc((size_t)NQROWS * 4);
    float*  meanb = (float*)alloc(HID * 4);
    float*  varb  = (float*)alloc(HID * 4);
    float*  s1    = (float*)alloc((size_t)BATCH * HID * 4);
    float*  s2    = (float*)alloc((size_t)BATCH * HID * 4);
    __bf16* wbf   = (__bf16*)alloc((size_t)HID * HID * 2);
    __bf16* prbf  = (__bf16*)alloc((size_t)NBP * DH * 2);

    auto gemm = [&](const __bf16* A, const __bf16* Bm, const float* bias,
                    float* C, __bf16* Cbf, int M, int K, int Nn, int lda, int ldb,
                    int batch, long sA, long sB, long sC, int flags) {
        if ((flags & GF_TRA) && (flags & GF_TRB)) {
            dim3 g(Nn / 32, M / 64, batch);
            k_gemm<true, true, 32><<<g, 128, 0, stream>>>(A, Bm, bias, C, Cbf, K, Nn, lda, ldb, sA, sB, sC, flags);
        } else if (flags & GF_TRB) {
            dim3 g(Nn / 32, M / 64, batch);
            k_gemm<false, true, 32><<<g, 128, 0, stream>>>(A, Bm, bias, C, Cbf, K, Nn, lda, ldb, sA, sB, sC, flags);
        } else {
            dim3 g(Nn / 64, M / 64, batch);   // all non-transposed calls have N % 64 == 0
            k_gemm<false, false, 64><<<g, 128, 0, stream>>>(A, Bm, bias, C, Cbf, K, Nn, lda, ldb, sA, sB, sC, flags);
        }
    };
    auto cast = [&](const float* src, __bf16* dst, int n) {
        k_cast<<<(n + 255) / 256, 256, 0, stream>>>(src, dst, n);
    };

    const int NE = NROWS * HID;            // 8.4M elementwise threads
    const float qk_scale = 0.42044820762685725f;  // DH^-0.25

    // ---------------- trunk ----------------
    k_embed<<<NE / 256, 256, 0, stream>>>(p, fc_p_W, fc_p_b, net);
    for (int i = 0; i < 5; ++i) {
        k_colstats<<<HID, 256, 0, stream>>>(net, meanb, varb, NROWS);
        k_cbn_params<<<8, 256, 0, stream>>>(cond, g0W + (long)i*HID*CDIM, g0b + i*HID,
                                            b0W + (long)i*HID*CDIM, b0b + i*HID,
                                            meanb, varb, s1, s2);
        k_cbn_relu<<<NE / 256, 256, 0, stream>>>(net, s1, s2, bf1);
        cast(f0W + (long)i*HID*HID, wbf, HID*HID);
        gemm(bf1, wbf, f0b + i*HID, hbuf, nullptr, NROWS, HID, HID, HID, HID,
             1, 0, 0, 0, GF_BIAS);

        k_colstats<<<HID, 256, 0, stream>>>(hbuf, meanb, varb, NROWS);
        k_cbn_params<<<8, 256, 0, stream>>>(cond, g1W + (long)i*HID*CDIM, g1b + i*HID,
                                            b1W + (long)i*HID*CDIM, b1b + i*HID,
                                            meanb, varb, s1, s2);
        k_cbn_relu<<<NE / 256, 256, 0, stream>>>(hbuf, s1, s2, bf1);
        cast(f1W + (long)i*HID*HID, wbf, HID*HID);
        gemm(bf1, wbf, f1b + i*HID, net, nullptr, NROWS, HID, HID, HID, HID,
             1, 0, 0, 0, GF_BIAS | GF_ACC);   // residual add
    }

    // proj_in: y = net @ proj_in_W^T + b
    cast(net, bf1, NE);
    cast(proj_in_W, wbf, PHID * HID);
    gemm(bf1, wbf, proj_in_b, ybuf, nullptr, NROWS, HID, PHID, HID, HID,
         1, 0, 0, 0, GF_BIAS);

    // ---------------- performer blocks ----------------
    float* qlin = hbuf;
    float* klin = hbuf + (size_t)NROWS * PHID;
    float* vlin = hbuf + 2 * (size_t)NROWS * PHID;
    const int NSPLIT = BHN * TLEN * DH;

    for (int l = 0; l < 3; ++l) {
        // --- attention ---
        k_ln<<<NROWS / 256, 256, 0, stream>>>(ybuf, ln1_s + l*PHID, ln1_b + l*PHID, lnbf);
        cast(Wq + (long)l*PHID*PHID, wbf, PHID*PHID);
        gemm(lnbf, wbf, nullptr, qlin, nullptr, NROWS, PHID, PHID, PHID, PHID, 1,0,0,0, 0);
        cast(Wk + (long)l*PHID*PHID, wbf, PHID*PHID);
        gemm(lnbf, wbf, nullptr, klin, nullptr, NROWS, PHID, PHID, PHID, PHID, 1,0,0,0, 0);
        cast(Wv + (long)l*PHID*PHID, wbf, PHID*PHID);
        gemm(lnbf, wbf, nullptr, vlin, nullptr, NROWS, PHID, PHID, PHID, PHID, 1,0,0,0, 0);

        k_split<<<NSPLIT / 256, 256, 0, stream>>>(qlin, xdq, qk_scale);
        k_split<<<NSPLIT / 256, 256, 0, stream>>>(klin, xdk, qk_scale);
        k_split<<<NSPLIT / 256, 256, 0, stream>>>(vlin, vbf, 1.0f);
        k_diag<<<NQROWS / 256, 256, 0, stream>>>(xdq, diagq);
        k_diag<<<NQROWS / 256, 256, 0, stream>>>(xdk, diagk);
        k_padproj<<<(NBP * DH) / 256, 256, 0, stream>>>(projW + (long)l*NBV*DH, prbf);

        // qp features: u = xd @ proj^T ; qp = favor(u, per-row max)
        gemm(xdq, prbf, nullptr, ubuf, nullptr, NQROWS, DH, NBP, DH, DH, 1,0,0,0, 0);
        k_rowmax<<<NQROWS / 256, 256, 0, stream>>>(ubuf, rmax);
        k_favor<<<(NQROWS * NBP) / 256, 256, 0, stream>>>(ubuf, diagq, rmax, 1, bf1);
        // kp features: per-(b,h) global max
        gemm(xdk, prbf, nullptr, ubuf, nullptr, NQROWS, DH, NBP, DH, DH, 1,0,0,0, 0);
        k_rowmax<<<NQROWS / 256, 256, 0, stream>>>(ubuf, rmax);
        k_bhmax<<<BHN, 256, 0, stream>>>(rmax, bhm);
        k_favor<<<(NQROWS * NBP) / 256, 256, 0, stream>>>(ubuf, diagk, bhm, 0, bf2);

        k_ksum<<<BHN, NBP, 0, stream>>>(bf2, ksum);
        // ctx[bh] = kp^T @ v   (M=128, K=T, N=32), batched over bh
        gemm(bf2, vbf, nullptr, ctx, nullptr, NBP, TLEN, DH, NBP, DH,
             BHN, (long)TLEN*NBP, (long)TLEN*DH, (long)NBP*DH, GF_TRA | GF_TRB);
        cast(ctx, ctxbf, BHN * NBP * DH);
        k_dinv<<<NQROWS / 256, 256, 0, stream>>>(bf1, ksum, dinv);
        // o[bh] = qp @ ctx  (M=T, K=128, N=32), batched
        gemm(bf1, ctxbf, nullptr, obuf, nullptr, TLEN, NBP, DH, NBP, DH,
             BHN, (long)TLEN*NBP, (long)NBP*DH, (long)TLEN*DH, GF_TRB);
        k_merge<<<(NROWS * PHID) / 256, 256, 0, stream>>>(obuf, dinv, omg);
        cast(Wo + (long)l*PHID*PHID, wbf, PHID*PHID);
        gemm(omg, wbf, bo + l*PHID, ybuf, nullptr, NROWS, PHID, PHID, PHID, PHID,
             1, 0, 0, 0, GF_BIAS | GF_ACC);   // y += attn out

        // --- FFN ---
        k_ln<<<NROWS / 256, 256, 0, stream>>>(ybuf, ln2_s + l*PHID, ln2_b + l*PHID, lnbf);
        cast(W1 + (long)l*4*PHID*PHID, wbf, 4*PHID*PHID);
        gemm(lnbf, wbf, b1p + l*4*PHID, ubuf, bf1, NROWS, PHID, 4*PHID, PHID, PHID,
             1, 0, 0, 0, GF_BIAS | GF_GELU);   // gelu -> bf16 secondary out
        cast(W2 + (long)l*PHID*4*PHID, wbf, 4*PHID*PHID);
        gemm(bf1, wbf, b2p + l*PHID, ybuf, nullptr, NROWS, 4*PHID, PHID, 4*PHID, 4*PHID,
             1, 0, 0, 0, GF_BIAS | GF_ACC);    // y += ffn out
    }

    // ---------------- head ----------------
    cast(ybuf, lnbf, NROWS * PHID);
    cast(proj_out_W, wbf, HID * PHID);
    gemm(lnbf, wbf, proj_out_b, net, nullptr, NROWS, PHID, HID, PHID, PHID,
         1, 0, 0, 0, GF_BIAS);
    k_fcout<<<NROWS / 256, 256, 0, stream>>>(net, fc_out_W, fc_out_b, out);
}
